// FactorizedSoftmaxV2_10273561772327
// MI455X (gfx1250) — matrix operand order repaired and checked by reading, not compiled, not run on hardware
//
#include <hip/hip_runtime.h>
#include <hip/hip_bf16.h>
#include <cstddef>

// ---------------- problem constants ----------------
#define N_TOKENS   4096
#define HIDDEN     1024
#define VOCAB      50257
#define CUT1       20000
#define CUT2       40000

#define MTILE      32                       // rows per workgroup
#define COLS_WG    512                      // columns per workgroup (8 waves x 64)
#define NGROUP     ((VOCAB + COLS_WG - 1) / COLS_WG)   // 99 column groups
#define VPAD       (NGROUP * COLS_WG)       // 50688 zero-padded vocab columns
#define KPHASE     512                      // K staged in two 32-row x 512 phases

typedef __attribute__((ext_vector_type(16))) __bf16 v16bf;
typedef __attribute__((ext_vector_type(8)))  float  v8f;

__device__ __forceinline__ int cluster_of(int v) {
    return v < CUT1 ? 0 : (v < CUT2 ? 1 : 2);
}

// xor-swap sum across the 16 lanes of each half-wave (single ds_swizzle per step)
__device__ __forceinline__ float half16_reduce(float v) {
    v += __int_as_float(__builtin_amdgcn_ds_swizzle(__float_as_int(v), 0x1F | (1 << 10)));
    v += __int_as_float(__builtin_amdgcn_ds_swizzle(__float_as_int(v), 0x1F | (2 << 10)));
    v += __int_as_float(__builtin_amdgcn_ds_swizzle(__float_as_int(v), 0x1F | (4 << 10)));
    v += __int_as_float(__builtin_amdgcn_ds_swizzle(__float_as_int(v), 0x1F | (8 << 10)));
    return v;
}

// ---------------- kernel 1: x fp32 -> bf16 ----------------
__global__ void cvt_x_kernel(const float* __restrict__ x, __bf16* __restrict__ xb, int n) {
    int i = blockIdx.x * blockDim.x + threadIdx.x;
    if (i < n) xb[i] = (__bf16)x[i];
}

// ---------------- kernel 2: logits fp32 [K=1024][V] -> bf16 transposed [VPAD][1024] ----
__global__ void transpose_cvt_kernel(const float* __restrict__ L, __bf16* __restrict__ Lt) {
    __shared__ float t[32][33];
    int c0 = blockIdx.x * 32;
    int k0 = blockIdx.y * 32;
    int tx = threadIdx.x, ty = threadIdx.y;        // blockDim (32,8)
#pragma unroll
    for (int i = 0; i < 4; ++i) {
        int k = k0 + ty + 8 * i;
        int c = c0 + tx;
        float v = 0.0f;
        if (c < VOCAB) v = L[(size_t)k * VOCAB + c];
        t[ty + 8 * i][tx] = v;
    }
    __syncthreads();
#pragma unroll
    for (int i = 0; i < 4; ++i) {
        int c = c0 + ty + 8 * i;
        int k = k0 + tx;
        Lt[(size_t)c * HIDDEN + k] = (__bf16)t[tx][ty + 8 * i];
    }
}

// ---------------- kernel 3: WMMA GEMM + fused masked exp-sum partials ----------------
// grid: (N_TOKENS/MTILE, NGROUP), block 256 (8 waves).
// Each wave: 32M x 64N macro-tile = 2x4 wmma tiles; K in two LDS-staged phases.
__global__ void __launch_bounds__(256)
gemm_lse_kernel(const __bf16* __restrict__ Xb, const __bf16* __restrict__ Lt,
                const int* __restrict__ y, float* __restrict__ Spart) {
    __shared__ __bf16 lx[MTILE * KPHASE];   // 32 KB x-tile (one K phase)
    __shared__ float  lws[8][MTILE];        // per-wave row sums
    __shared__ int    lcl[MTILE];           // row cluster ids

    const int tid = threadIdx.x;
    const int m0  = blockIdx.x * MTILE;

    if (tid < MTILE) lcl[tid] = cluster_of(y[m0 + tid]);

    const int wave = tid >> 5;
    const int lane = tid & 31;
    const int hf   = lane >> 4;        // half-wave: 0 or 1
    const int ln   = lane & 15;
    const int c0   = blockIdx.y * COLS_WG + wave * 64;

    // per-column-tile masks (column = c0 + ni*16 + ln for this lane)
    int  ccl[4];
    bool cval[4];
#pragma unroll
    for (int ni = 0; ni < 4; ++ni) {
        int col  = c0 + ni * 16 + ln;
        cval[ni] = (col < VOCAB);
        ccl[ni]  = cluster_of(cval[ni] ? col : VOCAB - 1);
    }

    // B base pointers (Lt zero-padded to VPAD columns -> always in-bounds)
    const __bf16* bbase[4];
#pragma unroll
    for (int ni = 0; ni < 4; ++ni)
        bbase[ni] = Lt + (size_t)(c0 + ni * 16 + ln) * HIDDEN + hf * 16;

    v8f acc[2][4];
#pragma unroll
    for (int mi = 0; mi < 2; ++mi)
#pragma unroll
        for (int ni = 0; ni < 4; ++ni)
            acc[mi][ni] = (v8f){0.f, 0.f, 0.f, 0.f, 0.f, 0.f, 0.f, 0.f};

    for (int ph = 0; ph < HIDDEN / KPHASE; ++ph) {
        __syncthreads();
        // stage 32 rows x 512 K of Xb into LDS (32 KB)
        {
            uint4* dst = reinterpret_cast<uint4*>(lx);
#pragma unroll
            for (int i = tid; i < MTILE * KPHASE / 8; i += 256) {
                int row = i >> 6;            // 64 uint4 per row
                int kc  = i & 63;
                dst[i] = *reinterpret_cast<const uint4*>(
                    Xb + (size_t)(m0 + row) * HIDDEN + ph * KPHASE + kc * 8);
            }
        }
        __syncthreads();

        const __bf16* pa0 = lx + ln * KPHASE + hf * 16;          // rows 0..15 (M=ln)
        const __bf16* pa1 = lx + (16 + ln) * KPHASE + hf * 16;   // rows 16..31

        for (int kk = 0; kk < KPHASE; kk += 32) {
            v16bf a[2], b[4];
#pragma unroll
            for (int e = 0; e < 8; ++e) {
                a[0][e] = pa0[kk + e];  a[0][8 + e] = pa0[kk + 8 + e];
                a[1][e] = pa1[kk + e];  a[1][8 + e] = pa1[kk + 8 + e];
            }
#pragma unroll
            for (int ni = 0; ni < 4; ++ni) {
                const __bf16* pb = bbase[ni] + ph * KPHASE + kk;
#pragma unroll
                for (int e = 0; e < 16; ++e) b[ni][e] = pb[e];
            }
#pragma unroll
            for (int mi = 0; mi < 2; ++mi)
#pragma unroll
                for (int ni = 0; ni < 4; ++ni)
                    acc[mi][ni] = __builtin_amdgcn_wmma_f32_16x16x32_bf16(
                        false, a[mi], false, b[ni], (short)0, acc[mi][ni], false, false);
        }
    }

    // Epilogue: masked exp, sum this wave's 64 columns per row, reduce across half-wave.
    // D layout: VGPR r -> row M = mi*16 + r + hf*8, col N = lane's column.
#pragma unroll
    for (int mi = 0; mi < 2; ++mi) {
#pragma unroll
        for (int r = 0; r < 8; ++r) {
            const int row = mi * 16 + r + hf * 8;
            const int rc  = lcl[row];
            float v = 0.0f;
#pragma unroll
            for (int ni = 0; ni < 4; ++ni)
                v += (cval[ni] && (ccl[ni] == rc)) ? __expf(acc[mi][ni][r]) : 0.0f;
            v = half16_reduce(v);
            if (ln == 0) lws[wave][row] = v;   // lane 0 / lane 16 each cover their rows
        }
    }
    __syncthreads();

    if (tid < MTILE) {   // deterministic ordered sum over the 8 waves
        float s = 0.0f;
#pragma unroll
        for (int w = 0; w < 8; ++w) s += lws[w][tid];
        Spart[(size_t)(m0 + tid) * NGROUP + blockIdx.y] = s;
    }
}

// ---------------- kernel 4: finalize per token ----------------
__global__ void __launch_bounds__(256)
finalize_kernel(const float* __restrict__ x, const int* __restrict__ y,
                const float* __restrict__ Wc, const __bf16* __restrict__ Lt,
                const float* __restrict__ Spart, float* __restrict__ out) {
    __shared__ float red[5][256];
    const int n = blockIdx.x;
    const int t = threadIdx.x;
    const int yv = y[n];

    const float*  xr = x  + (size_t)n  * HIDDEN;
    const __bf16* lr = Lt + (size_t)yv * HIDDEN;

    float p0 = 0.f, p1 = 0.f, p2 = 0.f, p3 = 0.f, p4 = 0.f;
#pragma unroll
    for (int k = t; k < HIDDEN; k += 256) {
        float xv = xr[k];
        p0 += xv * Wc[k];
        p1 += xv * Wc[HIDDEN + k];
        p2 += xv * Wc[2 * HIDDEN + k];
        p3 += xv * (float)lr[k];
    }
    for (int k = t; k < NGROUP; k += 256) p4 += Spart[(size_t)n * NGROUP + k];

    red[0][t] = p0; red[1][t] = p1; red[2][t] = p2; red[3][t] = p3; red[4][t] = p4;
    __syncthreads();
    for (int s = 128; s > 0; s >>= 1) {
        if (t < s) {
#pragma unroll
            for (int j = 0; j < 5; ++j) red[j][t] += red[j][t + s];
        }
        __syncthreads();
    }
    if (t == 0) {
        float z0 = red[0][0], z1 = red[1][0], z2 = red[2][0];
        float tgt = red[3][0], S = red[4][0];
        float m = fmaxf(z0, fmaxf(z1, z2));
        float lse3 = m + logf(__expf(z0 - m) + __expf(z1 - m) + __expf(z2 - m));
        int c = cluster_of(yv);
        float zc = (c == 0) ? z0 : ((c == 1) ? z1 : z2);
        out[n] = -(zc - lse3) - (tgt - logf(S));
    }
}

// ---------------- launcher ----------------
extern "C" void kernel_launch(void* const* d_in, const int* in_sizes, int n_in,
                              void* d_out, int out_size, void* d_ws, size_t ws_size,
                              hipStream_t stream) {
    const float* x  = (const float*)d_in[0];
    const int*   y  = (const int*)d_in[1];
    const float* Wc = (const float*)d_in[2];
    const float* L  = (const float*)d_in[3];
    float* out = (float*)d_out;

    // workspace layout
    char* ws = (char*)d_ws;
    __bf16* Xb = (__bf16*)ws;                                   //  8 MB
    size_t oLt = (size_t)N_TOKENS * HIDDEN * sizeof(__bf16);
    __bf16* Lt = (__bf16*)(ws + oLt);                           // ~104 MB (fits 192 MB L2)
    size_t oSp = oLt + (size_t)VPAD * HIDDEN * sizeof(__bf16);
    float* Spart = (float*)(ws + oSp);                          // 1.6 MB

    // 1) x -> bf16
    {
        int n = N_TOKENS * HIDDEN;
        cvt_x_kernel<<<(n + 255) / 256, 256, 0, stream>>>(x, Xb, n);
    }
    // 2) logits -> bf16 transposed, zero-padded to VPAD columns
    {
        dim3 g(VPAD / 32, HIDDEN / 32);
        dim3 b(32, 8);
        transpose_cvt_kernel<<<g, b, 0, stream>>>(L, Lt);
    }
    // 3) WMMA GEMM with fused masked exp-sum partials
    {
        dim3 g(N_TOKENS / MTILE, NGROUP);
        gemm_lse_kernel<<<g, 256, 0, stream>>>(Xb, Lt, y, Spart);
    }
    // 4) finalize NLL per token
    finalize_kernel<<<N_TOKENS, 256, 0, stream>>>(x, y, Wc, Lt, Spart, out);
}
